// adaptive_loss_74285754352024
// MI455X (gfx1250) — matrix operand order repaired and checked by reading, not compiled, not run on hardware
//
#include <hip/hip_runtime.h>
#include <cmath>

typedef __attribute__((ext_vector_type(16))) _Float16 v16h;
typedef __attribute__((ext_vector_type(8)))  float    v8f;

#define H_IMG 1024
#define W_IMG 1024
#define TILE  64
#define MR    96    // staged mask rows  (tile + 16 halo each side)
#define MC    112   // staged mask cols / stride (cols 0..111 used)
#define HS    112   // lds_H row-dim stride (rows 0..95 data, 96..111 zero pad)

__device__ __forceinline__ float bandv(int d, int p) {
    // box-filter band: input index - output index in [16-p, 16+p]
    return (d >= 16 - p && d <= 16 + p) ? 1.0f : 0.0f;
}

__global__ __launch_bounds__(256) void wloss_main(
    const float* __restrict__ pred, const float* __restrict__ mask,
    float* __restrict__ part)
{
    __shared__ _Float16 lmask[MR * MC];       // mask tile, f16, row-major
    __shared__ _Float16 lH[3][TILE * HS];     // horizontal sums, TRANSPOSED: [col][row]
    __shared__ float red[8][5];

    const int tid  = threadIdx.x;
    const int lane = tid & 31;
    const int wave = tid >> 5;
    const int img    = blockIdx.x >> 8;
    const int tileId = blockIdx.x & 255;
    const int y0 = (tileId >> 4) * TILE;
    const int x0 = (tileId & 15) * TILE;
    const size_t base = (size_t)img * (size_t)(H_IMG * W_IMG);

    // ---- per-lane constant band operands (shared by both passes) ----
    const int P[3] = {1, 7, 15};              // radii for k = 3, 15, 31
    v16h B0[3], B1[3];                        // B-layout (32x16), horizontal pass
    v16h S0[3], S1[3];                        // A-layout (16x32), vertical pass
    {
        const int i     = lane & 15;
        const int kbOff = (lane < 16) ? 0 : 16;   // S0/S1 (vertical A): lanes 0-15 K=0..15, 16-31 K=16..31
        const int kaOff = (lane < 16) ? 0 : 8;    // B0/B1 (horizontal B): lane halves offset K by 8
#pragma unroll
        for (int k = 0; k < 3; ++k) {
#pragma unroll
            for (int j = 0; j < 8; ++j)
#pragma unroll
                for (int h = 0; h < 2; ++h) {
                    int e  = 2 * j + h;
                    int kb = ((j < 4) ? 2 * j : 8 + 2 * j) + kaOff + h;
                    int ka = 2 * j + h + kbOff;
                    B0[k][e] = (_Float16)bandv(kb - i,      P[k]);
                    B1[k][e] = (_Float16)bandv(kb + 32 - i, P[k]);
                    S0[k][e] = (_Float16)bandv(ka - i,      P[k]);
                    S1[k][e] = (_Float16)bandv(ka + 32 - i, P[k]);
                }
        }
    }

    // ---- phase 1: stage mask tile with zero halo; zero lH pad rows ----
    for (int t = tid; t < MR * MC; t += 256) {
        int mr = t / MC, mc = t % MC;
        int gy = y0 - 16 + mr, gx = x0 - 16 + mc;
        float v = 0.0f;
        if (gy >= 0 && gy < H_IMG && gx >= 0 && gx < W_IMG)
            v = mask[base + (size_t)gy * W_IMG + gx];
        lmask[mr * MC + mc] = (_Float16)v;
    }
    for (int t = tid; t < 3 * TILE * 16; t += 256) {
        int k = t >> 10;
        int r = t & 1023;
        int col = r >> 4, row = 96 + (r & 15);
        lH[k][col * HS + row] = (_Float16)0.0f;
    }
    if (tid < 128) {  // pull pred tile toward L2 while we filter
        int ro = tid >> 1, co = (tid & 1) * 32;
        __builtin_prefetch(&pred[base + (size_t)(y0 + ro) * W_IMG + x0 + co], 0, 0);
    }
    __syncthreads();

    // ---- phase 2: horizontal box sums via WMMA -> lH (transposed) ----
    for (int b = wave; b < 24; b += 8) {
        const int rb = (b >> 2) * 16;     // H-row block base (0..80), tile-row coords
        const int cb = (b & 3) * 16;      // output-col block base (0..48)
        const int row   = rb + (lane & 15);
        const int kaOff = (lane < 16) ? 0 : 8;
        v16h a0, a1;                      // mask chunks in A-layout
#pragma unroll
        for (int j = 0; j < 8; ++j)
#pragma unroll
            for (int h = 0; h < 2; ++h) {
                int kl = ((j < 4) ? 2 * j : 8 + 2 * j) + kaOff + h;
                a0[2 * j + h] = lmask[row * MC + cb + kl];
                a1[2 * j + h] = lmask[row * MC + cb + 32 + kl];
            }
        const int Nv    = lane & 15;
        const int mBase = (lane < 16) ? 0 : 8;
#pragma unroll
        for (int k = 0; k < 3; ++k) {
            v8f c = {};
            c = __builtin_amdgcn_wmma_f32_16x16x32_f16(false, a0, false, B0[k],
                                                       (short)0, c, false, false);
            c = __builtin_amdgcn_wmma_f32_16x16x32_f16(false, a1, false, B1[k],
                                                       (short)0, c, false, false);
#pragma unroll
            for (int j = 0; j < 8; ++j)   // C layout: (M = mBase+j, N = Nv)
                lH[k][(cb + Nv) * HS + (rb + mBase + j)] = (_Float16)c[j];
        }
    }
    __syncthreads();

    // ---- phase 3: vertical box sums via WMMA, fused loss + reduction ----
    float sw = 0.f, swb = 0.f, sit = 0.f, sun = 0.f, sma = 0.f;
    const float inv9 = 1.0f / 9.0f, inv225 = 1.0f / 225.0f, inv961 = 1.0f / 961.0f;

    for (int b = wave; b < 16; b += 8) {
        const int ob = (b >> 2) * 16;     // output row block (0..48)
        const int xb = (b & 3) * 16;      // output col block (0..48)
        const int col   = xb + (lane & 15);
        const int kbOff = (lane < 16) ? 0 : 16;
        v8f C[3];
#pragma unroll
        for (int k = 0; k < 3; ++k) {
            v16h b0, b1;                  // H columns in B-layout (contiguous runs)
#pragma unroll
            for (int e = 0; e < 16; ++e) {
                b0[e] = lH[k][col * HS + ob + e + kbOff];
                b1[e] = lH[k][col * HS + ob + 32 + e + kbOff];
            }
            v8f c = {};
            c = __builtin_amdgcn_wmma_f32_16x16x32_f16(false, S0[k], false, b0,
                                                       (short)0, c, false, false);
            c = __builtin_amdgcn_wmma_f32_16x16x32_f16(false, S1[k], false, b1,
                                                       (short)0, c, false, false);
            C[k] = c;
        }
        const int mBase = (lane < 16) ? 0 : 8;
#pragma unroll
        for (int j = 0; j < 8; ++j) {
            int yo = ob + mBase + j;
            size_t gidx = base + (size_t)(y0 + yo) * W_IMG + (x0 + col);
            float m  = mask[gidx];        // f32, mostly L2 hits
            float pr = pred[gidx];
            float a3  = fabsf(C[0][j] * inv9   - m);
            float a15 = fabsf(C[1][j] * inv225 - m);
            float a31 = fabsf(C[2][j] * inv961 - m);
            float w = 1.0f + 5.0f * (a3 + a15 + a31);
            float bce = fmaxf(pr, 0.0f) - pr * m + log1pf(__expf(-fabsf(pr)));
            float p = 1.0f / (1.0f + __expf(-pr));
            sw  += w;
            swb += w * bce;
            sit += p * m * w;
            sun += (p + m) * w;
            sma += fabsf(p - m);
        }
    }

    // wave then block reduction (deterministic, no atomics)
#pragma unroll
    for (int off = 16; off > 0; off >>= 1) {
        sw  += __shfl_down(sw,  off, 32);
        swb += __shfl_down(swb, off, 32);
        sit += __shfl_down(sit, off, 32);
        sun += __shfl_down(sun, off, 32);
        sma += __shfl_down(sma, off, 32);
    }
    if (lane == 0) {
        red[wave][0] = sw;  red[wave][1] = swb; red[wave][2] = sit;
        red[wave][3] = sun; red[wave][4] = sma;
    }
    __syncthreads();
    if (tid < 5) {
        float s = 0.f;
#pragma unroll
        for (int w = 0; w < 8; ++w) s += red[w][tid];
        part[(size_t)blockIdx.x * 5 + tid] = s;
    }
}

__global__ __launch_bounds__(256) void wloss_final(
    const float* __restrict__ part, float* __restrict__ out)
{
    __shared__ float acc[256][5];
    __shared__ float simg[16][5];
    const int tid  = threadIdx.x;
    const int img  = tid >> 4, slot = tid & 15;
    float a[5] = {0.f, 0.f, 0.f, 0.f, 0.f};
    for (int s = 0; s < 16; ++s) {
        const float* p = &part[((size_t)img * 256 + slot + 16 * s) * 5];
#pragma unroll
        for (int c = 0; c < 5; ++c) a[c] += p[c];
    }
#pragma unroll
    for (int c = 0; c < 5; ++c) acc[tid][c] = a[c];
    __syncthreads();
    if (tid < 16) {
        float s[5] = {0.f, 0.f, 0.f, 0.f, 0.f};
        for (int u = 0; u < 16; ++u)
#pragma unroll
            for (int c = 0; c < 5; ++c) s[c] += acc[tid * 16 + u][c];
#pragma unroll
        for (int c = 0; c < 5; ++c) simg[tid][c] = s[c];
    }
    __syncthreads();
    if (tid == 0) {
        float tot_mae = 0.f;
        for (int i = 0; i < 16; ++i) tot_mae += simg[i][4];
        const float N = (float)(H_IMG * W_IMG);
        const float mae = tot_mae / (16.0f * N);
        float loss = 0.f;
        for (int i = 0; i < 16; ++i) {
            float Ws = simg[i][0], WB = simg[i][1], IT = simg[i][2], UN = simg[i][3];
            float wbce = WB / Ws;
            float wiou = 1.0f - (IT + 1.0f) / (UN - IT + 1.0f);
            float wmae = mae * Ws / (Ws - N);   // (sum w * mae) / sum(w - 1)
            loss += 0.7f * (wbce + wiou + wmae);
        }
        out[0] = loss / 16.0f;
    }
}

extern "C" void kernel_launch(void* const* d_in, const int* in_sizes, int n_in,
                              void* d_out, int out_size, void* d_ws, size_t ws_size,
                              hipStream_t stream) {
    const float* pred = (const float*)d_in[0];
    const float* mask = (const float*)d_in[1];
    float* part = (float*)d_ws;   // 16*256 blocks * 5 floats = 80 KB
    (void)in_sizes; (void)n_in; (void)out_size; (void)ws_size;
    wloss_main<<<dim3(16 * 256), dim3(256), 0, stream>>>(pred, mask, part);
    wloss_final<<<dim3(1), dim3(256), 0, stream>>>(part, (float*)d_out);
}
